// XLSTMMemoryCell_45724221834037
// MI455X (gfx1250) — matrix operand order repaired and checked by reading, not compiled, not run on hardware
//
#include <hip/hip_runtime.h>
#include <math.h>

// ---------------------------------------------------------------------------
// xLSTM memory cell for MI455X (gfx1250, wave32, WMMA).
//   B=64, T=512, D=256, U=512, M=64, NCOLS = 4U+M = 2112
// Grid: 48 persistent blocks = 4 row-groups (16 batch rows each, = WMMA M tile)
//       x 12 col-groups (11 N-tiles of 16 cols each). 256 thr = 8 waves/block.
// Per step: bf16 WMMA GEMM z = [x_t|h] @ [Wcat;Ucat] + b  (K = 256 + 512,
// i.e. 24 chained v_wmma_f32_16x16x32_bf16 per output tile, with the B
// fragments software-pipelined 2 deep to hide L2 latency), then
// gates/softmax/ring-buffer memory read and the h update. Two device-scope
// barriers per step via one-shot per-(rg,t) atomic counters in workspace.
// A-operands staged in LDS via gfx1250 async global->LDS (ASYNCcnt +
// s_wait_asynccnt); B-operands stream from L2-resident bf16 weights.
// ---------------------------------------------------------------------------

#define T_STEPS 512
#define BATCH   64
#define DDIM    256
#define UDIM    512
#define MDIM    64
#define NCOLS   2112
#define RG_ROWS 16
#define NRG     4
#define NCG     12
#define TILES_PER_CG 11
#define NBLOCKS (NRG * NCG)
#define NTHREADS 256
#define KFRAGS   ((DDIM + UDIM) / 32)   // 24 WMMA K-steps per tile

typedef __bf16 v16bf __attribute__((ext_vector_type(16)));
typedef float  v8f   __attribute__((ext_vector_type(8)));
typedef unsigned int u32x4 __attribute__((ext_vector_type(4)));

// int4 vector matching the async-to-LDS builtin's parameter type, plus
// address-space-qualified pointer typedefs (global = AS1, LDS = AS3).
typedef int v4i __attribute__((vector_size(16)));
typedef __attribute__((address_space(1))) v4i* gptr_v4i;
typedef __attribute__((address_space(3))) v4i* lptr_v4i;

union FragAB { v16bf v; u32x4 q[2]; };
union FragC  { v8f  v; float f[8]; };

// ---- workspace layout (bytes) ----
#define OFF_UB    ((size_t)0)                       // 2112*512 bf16  = 2,162,688
#define OFF_WB    ((size_t)2162688)                 // 2112*256 bf16  = 1,081,344
#define OFF_BC    ((size_t)3244032)                 // 2112 f32       =     8,448
#define OFF_XB    ((size_t)3252480)                 // 512*64*256 bf16= 16,777,216
#define OFF_ZB    ((size_t)20029696)                // 4*16*2112 f32  =   540,672
#define OFF_HB    ((size_t)20570368)                // 4*16*512 bf16  =    65,536
#define OFF_RING  ((size_t)20635904)                // 4*64*16*512 f32= 8,388,608
#define OFF_C1    ((size_t)29024512)                // 4*512 u32      =     8,192
#define OFF_C2    ((size_t)29032704)                // 4*512 u32      =     8,192
#define STATE_WORDS ((size_t)((65536 + 8388608 + 8192 + 8192) / 4))  // zeroed

__device__ __forceinline__ unsigned short f2bf(float x) {
  unsigned u = __float_as_uint(x);
  unsigned r = ((u >> 16) & 1u) + 0x7fffu;   // round-to-nearest-even
  return (unsigned short)((u + r) >> 16);
}

// ---- WMMA fragment loads (wave32, 16-bit layouts per CDNA5 ISA 7.12.2) ----
// A (16x32 MxK): lanes 0-15 row=lane, K in {0..7}U{16..23}; lanes 16-31
// row=lane-16, K in {8..15}U{24..31}. Two ds_load_b128 per fragment.
__device__ __forceinline__ v16bf load_a_frag(const unsigned short* lds, int K, int kb) {
  int lane = threadIdx.x & 31;
  int row  = lane & 15;
  int koff = (lane >> 4) << 3;            // 0 or 8
  int base = row * K + kb * 32;
  FragAB f;
  f.q[0] = *(const u32x4*)(lds + base + koff);
  f.q[1] = *(const u32x4*)(lds + base + 16 + koff);
  return f.v;
}
// B (32x16 KxN): lanes 0-15 col=lane, K in {0..7}U{16..23}; lanes 16-31 col=lane-16,
// K in {8..15}U{24..31}.  Weights packed column(n)-major so K-run is contiguous:
// two global_load_b128 per fragment, served from L2-resident weights.
__device__ __forceinline__ v16bf load_b_frag(const unsigned short* gcol, int K, int kb) {
  int lane  = threadIdx.x & 31;
  int col   = lane & 15;
  int khalf = (lane >> 4) << 3;           // 0 or 8
  const unsigned short* p = gcol + (size_t)col * K + kb * 32 + khalf;
  FragAB f;
  f.q[0] = *(const u32x4*)(p);
  f.q[1] = *(const u32x4*)(p + 16);
  return f.v;
}
// Combined K index: kb 0..7 -> x part (Wb, K=256), kb 8..23 -> h part (Ub, K=512)
__device__ __forceinline__ v16bf load_b_k(const unsigned short* wcol,
                                          const unsigned short* ucol, int kb) {
  return (kb < 8) ? load_b_frag(wcol, DDIM, kb) : load_b_frag(ucol, UDIM, kb - 8);
}
__device__ __forceinline__ v16bf load_a_k(const unsigned short* ldsX,
                                          const unsigned short* ldsH, int kb) {
  return (kb < 8) ? load_a_frag(ldsX, DDIM, kb) : load_a_frag(ldsH, UDIM, kb - 8);
}

// ---- device-scope one-shot barrier over the 12 col-blocks of a row-group ----
__device__ __forceinline__ void rg_barrier(unsigned* cnt, unsigned target) {
  __threadfence();           // release: make z / h / ring stores visible
  __syncthreads();
  if (threadIdx.x == 0) {
    atomicAdd(cnt, 1u);
    while (atomicAdd(cnt, 0u) < target) __builtin_amdgcn_s_sleep(2);
    __threadfence();         // acquire: invalidate near caches before reads
  }
  __syncthreads();
}

// ---- gfx1250 async global->LDS path, guarded so compile never fails ----
#if defined(__HIP_DEVICE_COMPILE__) && \
    __has_builtin(__builtin_amdgcn_global_load_async_to_lds_b128) && \
    __has_builtin(__builtin_amdgcn_s_wait_asynccnt)
#define USE_ASYNC_LDS 1
#else
#define USE_ASYNC_LDS 0
#endif

// ===========================================================================
// Pack kernels (fully parallel, run once per launch)
// ===========================================================================
struct WParams {
  const float* W[5];   // Wi,Wf,Wo,Wc,We   [D x ncol]
  const float* U[5];   // Ui,Uf,Uo,Uc,Ue   [U x ncol]
  const float* B[5];   // bi,bf,bo,bc,be   [ncol]
};

__global__ void pack_weights(WParams p, unsigned short* __restrict__ Ub,
                             unsigned short* __restrict__ Wb,
                             float* __restrict__ bcat) {
  int n = blockIdx.x;                 // 0..2111
  int seg = n >> 9;                   // 0..3, and 4 for n>=2048
  int col = n & 511;                  // also correct for seg 4 (0..63)
  int ncol = (seg == 4) ? MDIM : UDIM;
  const float* Us = p.U[seg];
  const float* Ws = p.W[seg];
  for (int k = threadIdx.x; k < UDIM; k += blockDim.x)
    Ub[(size_t)n * UDIM + k] = f2bf(Us[(size_t)k * ncol + col]);
  for (int k = threadIdx.x; k < DDIM; k += blockDim.x)
    Wb[(size_t)n * DDIM + k] = f2bf(Ws[(size_t)k * ncol + col]);
  if (threadIdx.x == 0) bcat[n] = p.B[seg][col];
}

__global__ void pack_x(const float* __restrict__ x, unsigned short* __restrict__ Xb) {
  int bid = blockIdx.x;               // enumerates (b,t)
  int b = bid >> 9, t = bid & 511;
  int d = threadIdx.x;                // 256 threads == DDIM
  Xb[((size_t)t * BATCH + b) * DDIM + d] =
      f2bf(x[((size_t)b * T_STEPS + t) * DDIM + d]);
}

__global__ void zero_words(unsigned* __restrict__ p, size_t n) {
  size_t i = (size_t)blockIdx.x * blockDim.x + threadIdx.x;
  if (i < n) p[i] = 0u;
}

// ===========================================================================
// Persistent recurrent kernel
// ===========================================================================
__global__ __launch_bounds__(NTHREADS) void xlstm_recurrent(
    const unsigned short* __restrict__ Ub,   // [2112][512] bf16, n-major
    const unsigned short* __restrict__ Wb,   // [2112][256] bf16, n-major
    const float* __restrict__ bcat,          // [2112]
    const unsigned short* __restrict__ Xb,   // [T][B][D] bf16
    float* __restrict__ Zbuf,                // [NRG][16][2112]
    unsigned short* __restrict__ Hb,         // [NRG][16][512] bf16
    float* __restrict__ ring,                // [NRG][64][16][512]
    unsigned* __restrict__ cnt1,             // [NRG][T]
    unsigned* __restrict__ cnt2,             // [NRG][T]
    float* __restrict__ out)                 // [B][T][U]
{
  __shared__ __align__(32) unsigned short ldsH[RG_ROWS * UDIM];  // 16 KB
  __shared__ __align__(32) unsigned short ldsX[RG_ROWS * DDIM];  //  8 KB
  __shared__ float ldsE[RG_ROWS * MDIM];                         //  4 KB
  __shared__ float ldsEinv[RG_ROWS];

  const int rg   = blockIdx.x / NCG;
  const int cg   = blockIdx.x % NCG;
  const int tid  = threadIdx.x;
  const int wave = tid >> 5;
  const int lane = tid & 31;

  float*          Zrg    = Zbuf + (size_t)rg * RG_ROWS * NCOLS;
  unsigned short* Hrg    = Hb   + (size_t)rg * RG_ROWS * UDIM;
  float*          ringrg = ring + (size_t)rg * 64 * RG_ROWS * UDIM;
  unsigned*       c1     = cnt1 + rg * T_STEPS;
  unsigned*       c2     = cnt2 + rg * T_STEPS;
  const int nj = ((UDIM - 1 - cg) / NCG) + 1;   // #u-columns this block owns

  for (int t = 0; t < T_STEPS; ++t) {
    // ---------- stage A operands (h_{t-1}, x_t) into LDS ----------
    const unsigned short* xsrc = Xb + ((size_t)t * BATCH + rg * RG_ROWS) * DDIM;
#if USE_ASYNC_LDS
    for (int c = tid; c < (RG_ROWS * UDIM) / 8; c += NTHREADS)
      __builtin_amdgcn_global_load_async_to_lds_b128(
          (gptr_v4i)(Hrg + 8 * c), (lptr_v4i)(&ldsH[8 * c]), 0, 0);
    for (int c = tid; c < (RG_ROWS * DDIM) / 8; c += NTHREADS)
      __builtin_amdgcn_global_load_async_to_lds_b128(
          (gptr_v4i)(xsrc + 8 * c), (lptr_v4i)(&ldsX[8 * c]), 0, 0);
    __builtin_amdgcn_s_wait_asynccnt(0);
#else
    for (int c = tid; c < (RG_ROWS * UDIM) / 8; c += NTHREADS)
      *(u32x4*)(ldsH + 8 * c) = *(const u32x4*)(Hrg + 8 * c);
    for (int c = tid; c < (RG_ROWS * DDIM) / 8; c += NTHREADS)
      *(u32x4*)(ldsX + 8 * c) = *(const u32x4*)(xsrc + 8 * c);
#endif
    __syncthreads();

    // ---------- phase 1: z-tile GEMM via bf16 WMMA, B pipelined 2 deep ------
    for (int ti = wave; ti < TILES_PER_CG; ti += (NTHREADS / 32)) {
      int n0 = (cg * TILES_PER_CG + ti) * 16;
      const unsigned short* wcol = Wb + (size_t)n0 * DDIM;
      const unsigned short* ucol = Ub + (size_t)n0 * UDIM;

      FragC acc;
      float bv = bcat[n0 + (lane & 15)];
#pragma unroll
      for (int r = 0; r < 8; ++r) acc.f[r] = bv;

      FragAB bb[2];
      bb[0].v = load_b_k(wcol, ucol, 0);      // prefetch K-step 0
      bb[1].v = load_b_k(wcol, ucol, 1);      // prefetch K-step 1
#pragma unroll
      for (int kb = 0; kb < KFRAGS; ++kb) {
        v16bf a = load_a_k(ldsX, ldsH, kb);
        acc.v = __builtin_amdgcn_wmma_f32_16x16x32_bf16(
            false, a, false, bb[kb & 1].v, (short)0, acc.v, false, false);
        if (kb + 2 < KFRAGS)                  // refill the slot just consumed
          bb[kb & 1].v = load_b_k(wcol, ucol, kb + 2);
      }

      // D layout: VGPR r -> row = r + 8*(lane>=16), col = lane&15
      int rbase = (lane >> 4) << 3;
      int colg  = n0 + (lane & 15);
#pragma unroll
      for (int r = 0; r < 8; ++r)
        Zrg[(size_t)(rbase + r) * NCOLS + colg] = acc.f[r];
    }

    rg_barrier(c1 + t, NCG);   // z complete for all 2112 columns

    // ---------- phase 2: softmax + gates + ring-buffer memory ----------
    if (tid < RG_ROWS) {
      int row = tid;
      const float* zr = Zrg + (size_t)row * NCOLS + 4 * UDIM;
      float mx = zr[0];
      for (int m = 1; m < MDIM; ++m) mx = fmaxf(mx, zr[m]);
      float s = 0.f;
      for (int m = 0; m < MDIM; ++m) {
        float e = expf(zr[m] - mx);
        ldsE[row * MDIM + m] = e;
        s += e;
      }
      ldsEinv[row] = 1.0f / s;
    }
    __syncthreads();

    for (int idx = tid; idx < RG_ROWS * nj; idx += NTHREADS) {
      int row = idx & 15;
      int j   = idx >> 4;
      int u   = cg + NCG * j;
      const float* zrow = Zrg + (size_t)row * NCOLS;
      float zi = zrow[u];
      float zf = zrow[UDIM + u];
      float zo = zrow[2 * UDIM + u];
      float zc = zrow[3 * UDIM + u];
      float ig = 1.f / (1.f + expf(-zi));
      float fg = 1.f / (1.f + expf(-zf));
      float og = 1.f / (1.f + expf(-zo));
      float ct = tanhf(zc);

      float mc = 0.f;
      for (int m = 0; m < MDIM; ++m) {
        int slot = (t - 1 - m) & 63;   // ring: slot p holds c_p; unwritten = 0
        mc += ldsE[row * MDIM + m] *
              ringrg[((size_t)slot * RG_ROWS + row) * UDIM + u];
      }
      mc *= ldsEinv[row];

      float c = fg * mc + ig * ct;
      ringrg[((size_t)(t & 63) * RG_ROWS + row) * UDIM + u] = c;
      float h = og * tanhf(c);
      int b = rg * RG_ROWS + row;
      out[((size_t)b * T_STEPS + t) * UDIM + u] = h;
      Hrg[row * UDIM + u] = f2bf(h);
    }

    rg_barrier(c2 + t, NCG);   // h / ring complete before next step's GEMM
  }
}

// ===========================================================================
extern "C" void kernel_launch(void* const* d_in, const int* in_sizes, int n_in,
                              void* d_out, int out_size, void* d_ws, size_t ws_size,
                              hipStream_t stream) {
  (void)in_sizes; (void)n_in; (void)out_size; (void)ws_size;

  // setup_inputs order: x, Wi, Ui, bi, Wf, Uf, bf, Wo, Uo, bo, Wc, Uc, bc, We, Ue, be
  WParams p;
  p.W[0] = (const float*)d_in[1];  p.U[0] = (const float*)d_in[2];  p.B[0] = (const float*)d_in[3];
  p.W[1] = (const float*)d_in[4];  p.U[1] = (const float*)d_in[5];  p.B[1] = (const float*)d_in[6];
  p.W[2] = (const float*)d_in[7];  p.U[2] = (const float*)d_in[8];  p.B[2] = (const float*)d_in[9];
  p.W[3] = (const float*)d_in[10]; p.U[3] = (const float*)d_in[11]; p.B[3] = (const float*)d_in[12];
  p.W[4] = (const float*)d_in[13]; p.U[4] = (const float*)d_in[14]; p.B[4] = (const float*)d_in[15];
  const float* x = (const float*)d_in[0];

  char* ws = (char*)d_ws;
  unsigned short* Ub   = (unsigned short*)(ws + OFF_UB);
  unsigned short* Wb   = (unsigned short*)(ws + OFF_WB);
  float*          bcat = (float*)(ws + OFF_BC);
  unsigned short* Xb   = (unsigned short*)(ws + OFF_XB);
  float*          Zb   = (float*)(ws + OFF_ZB);
  unsigned short* Hb   = (unsigned short*)(ws + OFF_HB);
  float*          ring = (float*)(ws + OFF_RING);
  unsigned*       c1   = (unsigned*)(ws + OFF_C1);
  unsigned*       c2   = (unsigned*)(ws + OFF_C2);

  pack_weights<<<dim3(NCOLS), dim3(256), 0, stream>>>(p, Ub, Wb, bcat);
  pack_x<<<dim3(BATCH * T_STEPS), dim3(DDIM), 0, stream>>>(x, Xb);
  zero_words<<<dim3((unsigned)((STATE_WORDS + 255) / 256)), dim3(256), 0, stream>>>(
      (unsigned*)(ws + OFF_HB), STATE_WORDS);
  xlstm_recurrent<<<dim3(NBLOCKS), dim3(NTHREADS), 0, stream>>>(
      Ub, Wb, bcat, Xb, Zb, Hb, ring, c1, c2, (float*)d_out);
}
